// LSTMModel_64596308132529
// MI455X (gfx1250) — matrix operand order repaired and checked by reading, not compiled, not run on hardware
//
#include <hip/hip_runtime.h>

typedef __attribute__((ext_vector_type(16))) _Float16 v16h;
typedef __attribute__((ext_vector_type(8)))  _Float16 v8h;
typedef __attribute__((ext_vector_type(8)))  float    v8f;

#define HID   50            // hidden size
#define G4    200           // 4*HID gate width
#define TLEN  1024          // timesteps
#define NT    13            // N tiles of 16 covering 200 (pad to 208)
#define NTHREADS (NT * 32)  // 416 threads = 13 wave32s
#define CHUNK 128           // x timesteps staged per LDS refill
#define GPAD  208           // padded gate row stride (floats)

// Fast activations: v_exp_f32 + v_rcp_f32 (avoid IEEE div expansion; rcp is
// ~1ulp, far above the f16 h-state quantization noise floor).
__device__ __forceinline__ float sigmoid_f(float x) {
    return __builtin_amdgcn_rcpf(1.f + __expf(-x));
}
__device__ __forceinline__ float tanh_f(float x) {
    return __builtin_fmaf(2.f, __builtin_amdgcn_rcpf(1.f + __expf(-2.f * x)), -1.f);
}

__global__ __launch_bounds__(NTHREADS)
void lstm_wmma_kernel(const float* __restrict__ x,     // [B, T, 1]
                      const float* __restrict__ W_ih,  // [200, 1]
                      const float* __restrict__ W_hh,  // [200, 50]
                      const float* __restrict__ b_ih,  // [200]
                      const float* __restrict__ b_hh,  // [200]
                      const float* __restrict__ W_lin, // [1, 50]
                      const float* __restrict__ b_lin, // [1]
                      float* __restrict__ out)         // [B, 1]
{
    __shared__ __align__(16) _Float16 h_lds[16 * 64];   // h tile, K padded 50->64 (f16)
    __shared__ float gate_lds[16 * GPAD];               // post-GEMM gates (f32)
    __shared__ float x_lds[CHUNK * 16];                 // staged x, [tt][row]

    const int tid  = threadIdx.x;
    const int lane = tid & 31;
    const int wv   = tid >> 5;          // wave id 0..12 = N tile
    const int b0   = blockIdx.x * 16;   // batch row base

    // ---- per-lane constants for this wave's gate-column tile ----
    const int  ncol = wv * 16 + (lane & 15);   // 0..207 (200..207 = padding)
    const bool nval = (ncol < G4);
    const float wih_n  = nval ? W_ih[ncol] : 0.f;
    const float bsum_n = nval ? (b_ih[ncol] + b_hh[ncol]) : 0.f;

    // ---- preload B fragments of W_hh^T (f16), K padded to 64 (2 k-steps of 32)
    // 16-bit B layout: lanes 0-15 hold K=0..7,16..23, lanes 16-31 hold K=8..15,24..31
    v16h bf0, bf1;
    {
        const int kb = (lane < 16) ? 0 : 8;
        #pragma unroll
        for (int e = 0; e < 16; ++e) {
            const int k0 = kb + e + ((e >= 8) ? 8 : 0);  // < 32, always valid vs HID=50
            const int k1 = k0 + 32;         // may exceed 49 -> zero pad
            bf0[e] = nval ? (_Float16)W_hh[ncol * HID + k0] : (_Float16)0.f;
            bf1[e] = (nval && k1 < HID) ? (_Float16)W_hh[ncol * HID + k1] : (_Float16)0.f;
        }
    }

    // ---- zero h tile (also zeroes K-pad columns 50..63, which stay zero) ----
    for (int i = tid; i < 16 * 64; i += NTHREADS) h_lds[i] = (_Float16)0.f;

    // ---- register-resident cell state: 800 = 16 rows x 50 hidden cells ----
    const int e1 = tid;                  // always < 800
    const int e2 = tid + NTHREADS;
    const int m1 = e1 / HID, j1 = e1 % HID;
    const int m2 = e2 / HID, j2 = e2 % HID;
    const bool act2 = (e2 < 16 * HID);
    float c1 = 0.f, c2 = 0.f;

    // A fragment addressing (16-bit A layout): lane<16 -> K base 0, lane>=16 -> K base 8
    const int rb    = (lane < 16) ? 0 : 8;              // also C-row base
    const int abase = (lane & 15) * 64 + rb;            // halves offset into h_lds

    for (int t0 = 0; t0 < TLEN; t0 += CHUNK) {
        // stage x[b0..b0+15][t0..t0+CHUNK-1] into LDS, coalesced along t
        for (int idx = tid; idx < 16 * CHUNK; idx += NTHREADS) {
            const int m  = idx >> 7;            // idx / CHUNK
            const int tt = idx & (CHUNK - 1);
            x_lds[tt * 16 + m] = x[(size_t)(b0 + m) * TLEN + (t0 + tt)];
        }
        __syncthreads();

        for (int tt = 0; tt < CHUNK; ++tt) {
            // ---- load A fragments of h (two 16B LDS loads per 32-K fragment) ----
            v8h a0lo = *(const v8h*)&h_lds[abase];        // K kb+0..7
            v8h a0hi = *(const v8h*)&h_lds[abase + 16];   // K kb+16..23
            v8h a1lo = *(const v8h*)&h_lds[abase + 32];   // +32 k-step
            v8h a1hi = *(const v8h*)&h_lds[abase + 48];
            v16h a0 = __builtin_shufflevector(a0lo, a0hi, 0,1,2,3,4,5,6,7,8,9,10,11,12,13,14,15);
            v16h a1 = __builtin_shufflevector(a1lo, a1hi, 0,1,2,3,4,5,6,7,8,9,10,11,12,13,14,15);

            // ---- C init = x[b,t]*W_ih[n] + (b_ih+b_hh)[n]  (C layout: vgpr j -> row rb+j)
            const float* xr = &x_lds[tt * 16 + rb];
            v8f acc;
            #pragma unroll
            for (int j = 0; j < 8; ++j) acc[j] = xr[j] * wih_n + bsum_n;

            // ---- gates += h @ W_hh^T  (two k-steps of 32, f32 accumulate) ----
            acc = __builtin_amdgcn_wmma_f32_16x16x32_f16(false, a0, false, bf0,
                                                         (short)0, acc, false, false);
            acc = __builtin_amdgcn_wmma_f32_16x16x32_f16(false, a1, false, bf1,
                                                         (short)0, acc, false, false);

            // ---- scatter gate tile to LDS ----
            #pragma unroll
            for (int j = 0; j < 8; ++j) gate_lds[(rb + j) * GPAD + ncol] = acc[j];
            __syncthreads();

            // ---- elementwise LSTM cell update (gate order i,f,g,o) ----
            {
                const float gi = gate_lds[m1 * GPAD + j1];
                const float gf = gate_lds[m1 * GPAD + HID + j1];
                const float gg = gate_lds[m1 * GPAD + 2 * HID + j1];
                const float go = gate_lds[m1 * GPAD + 3 * HID + j1];
                c1 = sigmoid_f(gf) * c1 + sigmoid_f(gi) * tanh_f(gg);
                h_lds[m1 * 64 + j1] = (_Float16)(sigmoid_f(go) * tanh_f(c1));
            }
            if (act2) {
                const float gi = gate_lds[m2 * GPAD + j2];
                const float gf = gate_lds[m2 * GPAD + HID + j2];
                const float gg = gate_lds[m2 * GPAD + 2 * HID + j2];
                const float go = gate_lds[m2 * GPAD + 3 * HID + j2];
                c2 = sigmoid_f(gf) * c2 + sigmoid_f(gi) * tanh_f(gg);
                h_lds[m2 * 64 + j2] = (_Float16)(sigmoid_f(go) * tanh_f(c2));
            }
            __syncthreads();
        }
    }

    // ---- linear head on last h: out[b] = h . W_lin + b_lin ----
    if (tid < 16) {
        float acc = b_lin[0];
        #pragma unroll 10
        for (int j = 0; j < HID; ++j)
            acc += (float)h_lds[tid * 64 + j] * W_lin[j];
        out[b0 + tid] = acc;
    }
}

extern "C" void kernel_launch(void* const* d_in, const int* in_sizes, int n_in,
                              void* d_out, int out_size, void* d_ws, size_t ws_size,
                              hipStream_t stream) {
    const float* x     = (const float*)d_in[0];
    const float* W_ih  = (const float*)d_in[1];
    const float* W_hh  = (const float*)d_in[2];
    const float* b_ih  = (const float*)d_in[3];
    const float* b_hh  = (const float*)d_in[4];
    const float* W_lin = (const float*)d_in[5];
    const float* b_lin = (const float*)d_in[6];
    float* out = (float*)d_out;

    const int B = in_sizes[0] / TLEN;   // 2048
    lstm_wmma_kernel<<<B / 16, NTHREADS, 0, stream>>>(
        x, W_ih, W_hh, b_ih, b_hh, W_lin, b_lin, out);
}
